// GraphGNN_13554916786314
// MI455X (gfx1250) — matrix operand order repaired and checked by reading, not compiled, not run on hardware
//
#include <hip/hip_runtime.h>

// Problem constants (from reference)
#define Bc 4
#define Nc 50000
#define Ec 800000
// IN_DIM=16, E_H=32, E_OUT=30, OUT_DIM=13; concat feature = 35 (+1 bias) padded to 64

static_assert(Ec % 16 == 0, "tile tail handling removed; E must be divisible by 16");

typedef __attribute__((ext_vector_type(16))) __bf16 v16bf;
typedef __attribute__((ext_vector_type(8)))  float  v8f;

union FragBF {
    unsigned int u[8];
    v16bf v;
};

// Native RNE conversion (compiler emits packed v_cvt_pk_bf16_f32 where possible)
__device__ __forceinline__ unsigned short f32_to_bf16(float f) {
    __bf16 b = (__bf16)f;
    return __builtin_bit_cast(unsigned short, b);
}

// sigmoid via hardware v_exp_f32 + v_rcp_f32 (no IEEE division sequence)
__device__ __forceinline__ float sigmoidf_fast(float v) {
    return __builtin_amdgcn_rcpf(1.0f + __expf(-v));
}

// Intra-wave LDS ordering: all DS stores visible to subsequent DS loads
// (cross-lane communication within one wave32; no block barrier needed).
__device__ __forceinline__ void wave_lds_fence() {
    __builtin_amdgcn_wave_barrier();
    asm volatile("s_wait_dscnt 0" ::: "memory");
    __builtin_amdgcn_wave_barrier();
}

// ---------------------------------------------------------------------------
// Kernel 0: zero agg scratch
// ---------------------------------------------------------------------------
__global__ __launch_bounds__(256) void zero_kernel(float4* __restrict__ p, long long n4) {
    long long i = (long long)blockIdx.x * blockDim.x + threadIdx.x;
    long long stride = (long long)gridDim.x * blockDim.x;
    float4 z = make_float4(0.f, 0.f, 0.f, 0.f);
    for (; i < n4; i += stride) p[i] = z;
}

// ---------------------------------------------------------------------------
// Kernel 1: edge MLP (WMMA bf16) + scatter-add
// ---------------------------------------------------------------------------
__global__ __launch_bounds__(256) void edge_kernel(
    const float* __restrict__ x,        // (B, N, 16)
    const float* __restrict__ ew,       // (B, E, 3)
    const int*   __restrict__ eidx,     // (2, E)
    const float* __restrict__ W1,       // (35, 32)
    const float* __restrict__ b1,       // (32,)
    const float* __restrict__ W2,       // (32, 30)
    const float* __restrict__ b2,       // (30,)
    float*       __restrict__ agg)      // (B, N, 30) scratch
{
    __shared__ unsigned short sW1[64][32];      // K padded 35(+bias row 35)->64, bf16
    __shared__ unsigned short sW2[32][32];      // N padded 30->32, bf16
    __shared__ float sB2[32];
    __shared__ unsigned short sFeat[8][16][64]; // per-wave A tile (16 edges x K64)
    __shared__ unsigned short sH[8][16][32];    // per-wave hidden tile
    __shared__ int sSrcOff[8][16];              // src_idx * 30 (row offset into agg)
    __shared__ int sTgtOff[8][16];              // tgt_idx * 30

    const int tid = threadIdx.x;

    // ---- stage weights into LDS (bf16, padded; row 35 of W1 carries b1) ----
    for (int i = tid; i < 64 * 32; i += 256) {
        int k = i >> 5, n = i & 31;
        float v = (k < 35) ? W1[k * 32 + n] : ((k == 35) ? b1[n] : 0.0f);
        sW1[k][n] = f32_to_bf16(v);
    }
    for (int i = tid; i < 32 * 32; i += 256) {
        int k = i >> 5, n = i & 31;
        float v = (n < 30) ? W2[k * 30 + n] : 0.0f;
        sW2[k][n] = f32_to_bf16(v);
    }
    if (tid < 32) sB2[tid] = (tid < 30) ? b2[tid] : 0.0f;
    __syncthreads();

    const int lane = tid & 31;
    const int w    = tid >> 5;          // wave slab id
    const int col  = lane & 15;
    const int half = lane >> 4;         // half-wave id
    const int kb   = half * 16;         // K offset for B fragments (half-wave split)
    const int pb   = half * 8;          // unused; A fragments also use kb
    const int Mbase = half * 8;         // C-layout row base

    // ---- build B fragments, kept in VGPRs ----
    // B uses contiguous K order (A matches): lane(0-15)=col N, K=0..15 ; lane(16-31)=col N, K=16..31
    FragBF bw1[2][2];   // [N-tile][K-half]
    FragBF bw2[2];      // [N-tile]
#pragma unroll
    for (int t = 0; t < 2; ++t) {
        int c = col + 16 * t;
#pragma unroll
        for (int h = 0; h < 2; ++h) {
#pragma unroll
            for (int i = 0; i < 8; ++i) {
                int K = 32 * h + kb + 2 * i;
                bw1[t][h].u[i] = (unsigned)sW1[K][c] | ((unsigned)sW1[K + 1][c] << 16);
            }
        }
#pragma unroll
        for (int i = 0; i < 8; ++i) {
            int K = kb + 2 * i;
            bw2[t].u[i] = (unsigned)sW2[K][c] | ((unsigned)sW2[K + 1][c] << 16);
        }
    }

    // one-time constant region of the feature tile: K=35 is the bias input (1.0),
    // K=36..63 zero padding
    if (lane < 16) {
        sFeat[w][lane][35] = (unsigned short)0x3F80;  // bf16(1.0)
        for (int i = 36; i < 64; ++i) sFeat[w][lane][i] = 0;
    }
    wave_lds_fence();

    const int tilesPerB  = Ec / 16;
    const int totalTiles = Bc * tilesPerB;
    int wid     = ((int)(blockIdx.x * blockDim.x) + tid) >> 5;
    int wstride = (int)(gridDim.x * blockDim.x) >> 5;

    for (int tile = wid; tile < totalTiles; tile += wstride) {
        int b  = tile / tilesPerB;
        int e0 = (tile - b * tilesPerB) * 16;
        float* __restrict__ aggB = agg + (size_t)b * Nc * 30;

        // ---- gather (balanced across half-waves):
        //   lanes 0-15 : src node row -> K 0..15, src offset
        //   lanes 16-31: tgt node row -> K 16..31, edge weight -> K 32..34, tgt offset
        {
            int m = col;
            int e = e0 + m;
            if (half == 0) {
                int s = eidx[e];
                sSrcOff[w][m] = s * 30;
                const float4* xs = (const float4*)(x + ((size_t)b * Nc + s) * 16);
#pragma unroll
                for (int i = 0; i < 4; ++i) {
                    float4 v = xs[i];
                    sFeat[w][m][4 * i + 0] = f32_to_bf16(v.x);
                    sFeat[w][m][4 * i + 1] = f32_to_bf16(v.y);
                    sFeat[w][m][4 * i + 2] = f32_to_bf16(v.z);
                    sFeat[w][m][4 * i + 3] = f32_to_bf16(v.w);
                }
            } else {
                int t = eidx[Ec + e];
                sTgtOff[w][m] = t * 30;
                const float4* xt = (const float4*)(x + ((size_t)b * Nc + t) * 16);
#pragma unroll
                for (int i = 0; i < 4; ++i) {
                    float4 v = xt[i];
                    sFeat[w][m][16 + 4 * i + 0] = f32_to_bf16(v.x);
                    sFeat[w][m][16 + 4 * i + 1] = f32_to_bf16(v.y);
                    sFeat[w][m][16 + 4 * i + 2] = f32_to_bf16(v.z);
                    sFeat[w][m][16 + 4 * i + 3] = f32_to_bf16(v.w);
                }
                const float* wv = ew + ((size_t)b * Ec + e) * 3;
                sFeat[w][m][32] = f32_to_bf16(wv[0]);
                sFeat[w][m][33] = f32_to_bf16(wv[1]);
                sFeat[w][m][34] = f32_to_bf16(wv[2]);
            }
        }
        wave_lds_fence();

        // ---- load A fragments in contiguous K order matching B (lane=row, pairs of K) ----
        FragBF aLo, aHi;
        const unsigned int* featRow = (const unsigned int*)&sFeat[w][col][0];
#pragma unroll
        for (int i = 0; i < 8; ++i) {
            int k32 = kb + 2 * i; // even
            aLo.u[i] = featRow[k32 >> 1];
            aHi.u[i] = featRow[(32 + k32) >> 1];
        }

        // ---- layer 1: h = sigmoid(A(16x64) @ W1(64x32)), bias folded into K=35 ----
#pragma unroll
        for (int t = 0; t < 2; ++t) {
            v8f c = {};
            c = __builtin_amdgcn_wmma_f32_16x16x32_bf16(false, aLo.v, false, bw1[t][0].v,
                                                        (short)0, c, false, false);
            c = __builtin_amdgcn_wmma_f32_16x16x32_bf16(false, aHi.v, false, bw1[t][1].v,
                                                        (short)0, c, false, false);
#pragma unroll
            for (int r = 0; r < 8; ++r) {
                float hv = sigmoidf_fast(c[r]);
                sH[w][Mbase + r][col + 16 * t] = f32_to_bf16(hv);  // transpose via LDS
            }
        }
        wave_lds_fence();

        // ---- re-form A (16x32) for layer 2 ----
        FragBF a2;
        const unsigned int* hRow = (const unsigned int*)&sH[w][col][0];
#pragma unroll
        for (int i = 0; i < 8; ++i) {
            int k32 = kb + 2 * i;
            a2.u[i] = hRow[k32 >> 1];
        }

        // ---- vectorized fetch of the 8 row offsets this lane scatters to ----
        int soff[8], toff[8];
        {
            const int4* sp = (const int4*)&sSrcOff[w][Mbase];
            const int4* tp = (const int4*)&sTgtOff[w][Mbase];
            int4 sA = sp[0], sB4 = sp[1];
            int4 tA = tp[0], tB4 = tp[1];
            soff[0] = sA.x; soff[1] = sA.y; soff[2] = sA.z; soff[3] = sA.w;
            soff[4] = sB4.x; soff[5] = sB4.y; soff[6] = sB4.z; soff[7] = sB4.w;
            toff[0] = tA.x; toff[1] = tA.y; toff[2] = tA.z; toff[3] = tA.w;
            toff[4] = tB4.x; toff[5] = tB4.y; toff[6] = tB4.z; toff[7] = tB4.w;
        }

        // ---- layer 2 + scatter: e = sigmoid(h @ W2 + b2); agg[tgt]+=e; agg[src]-=e ----
#pragma unroll
        for (int t = 0; t < 2; ++t) {
            v8f c = {};
            c = __builtin_amdgcn_wmma_f32_16x16x32_bf16(false, a2.v, false, bw2[t].v,
                                                        (short)0, c, false, false);
            int n = col + 16 * t;
            if (n < 30) {
                float bias = sB2[n];
#pragma unroll
                for (int r = 0; r < 8; ++r) {
                    float evv = sigmoidf_fast(c[r] + bias);
                    float* pt = aggB + toff[r] + n;
                    float* ps = aggB + soff[r] + n;
                    __hip_atomic_fetch_add(pt,  evv, __ATOMIC_RELAXED, __HIP_MEMORY_SCOPE_AGENT);
                    __hip_atomic_fetch_add(ps, -evv, __ATOMIC_RELAXED, __HIP_MEMORY_SCOPE_AGENT);
                }
            }
        }
        wave_lds_fence();  // reads of sSrcOff/sTgtOff/sH done before next iter overwrites
    }
}

// ---------------------------------------------------------------------------
// Kernel 2: out = sigmoid(agg @ W3 + b3)   (B*N rows, 30 -> 13)
// ---------------------------------------------------------------------------
__global__ __launch_bounds__(256) void node_kernel(
    const float* __restrict__ agg,   // (B, N, 30)
    const float* __restrict__ W3,    // (30, 13)
    const float* __restrict__ b3,    // (13,)
    float*       __restrict__ out)   // (B, N, 13)
{
    __shared__ float sW3[30 * 13];
    __shared__ float sb3[13];
    for (int i = threadIdx.x; i < 30 * 13; i += 256) sW3[i] = W3[i];
    if (threadIdx.x < 13) sb3[threadIdx.x] = b3[threadIdx.x];
    __syncthreads();

    long long total  = (long long)Bc * Nc;
    long long i      = (long long)blockIdx.x * blockDim.x + threadIdx.x;
    long long stride = (long long)gridDim.x * blockDim.x;
    for (; i < total; i += stride) {
        const float* a = agg + i * 30;
        float acc[13];
#pragma unroll
        for (int j = 0; j < 13; ++j) acc[j] = sb3[j];
#pragma unroll
        for (int k = 0; k < 30; ++k) {
            float av = a[k];
#pragma unroll
            for (int j = 0; j < 13; ++j) acc[j] = fmaf(av, sW3[k * 13 + j], acc[j]);
        }
        float* o = out + i * 13;
#pragma unroll
        for (int j = 0; j < 13; ++j) o[j] = sigmoidf_fast(acc[j]);
    }
}

// ---------------------------------------------------------------------------
extern "C" void kernel_launch(void* const* d_in, const int* in_sizes, int n_in,
                              void* d_out, int out_size, void* d_ws, size_t ws_size,
                              hipStream_t stream) {
    (void)in_sizes; (void)n_in; (void)out_size; (void)ws_size;
    const float* x   = (const float*)d_in[0];
    const float* ew  = (const float*)d_in[1];
    const int*   ei  = (const int*)d_in[2];
    const float* W1  = (const float*)d_in[3];
    const float* b1  = (const float*)d_in[4];
    const float* W2  = (const float*)d_in[5];
    const float* b2  = (const float*)d_in[6];
    const float* W3  = (const float*)d_in[7];
    const float* b3  = (const float*)d_in[8];
    float* out = (float*)d_out;
    float* agg = (float*)d_ws;            // (B, N, 30) f32 = 24 MB

    long long aggN  = (long long)Bc * Nc * 30;
    long long aggN4 = aggN / 4;           // 6,000,000 divisible by 4
    zero_kernel<<<2048, 256, 0, stream>>>((float4*)agg, aggN4);

    edge_kernel<<<4096, 256, 0, stream>>>(x, ew, ei, W1, b1, W2, b2, agg);

    node_kernel<<<1024, 256, 0, stream>>>(agg, W3, b3, out);
}
